// TransformerEncoderLayer_40346922778679
// MI455X (gfx1250) — matrix operand order repaired and checked by reading, not compiled, not run on hardware
//
#include <hip/hip_runtime.h>
#include <hip/hip_bf16.h>

typedef __attribute__((ext_vector_type(16))) _Float16 v16h;
typedef __attribute__((ext_vector_type(8)))  _Float16 v8h;
typedef __attribute__((ext_vector_type(8)))  float    v8f;
typedef __attribute__((ext_vector_type(4)))  int      v4i;

#define D_MODEL  1024
#define N_HEAD   16
#define HEAD_DIM 64
#define D_FF     4096
#define BATCH    2
#define SEQ      2048
#define MROWS    (BATCH * SEQ)

#if __has_builtin(__builtin_amdgcn_global_load_async_to_lds_b128)
#define ASYNC_OK 1
#else
#define ASYNC_OK 0
#endif

// ---------------------------------------------------------------------------
// CDNA5 helpers
// ---------------------------------------------------------------------------
__device__ __forceinline__ v8f wmma_f16(v16h a, v16h b, v8f c) {
  return __builtin_amdgcn_wmma_f32_16x16x32_f16(
      /*neg_a=*/false, a, /*neg_b=*/false, b,
      /*c_mod=*/(short)0, c, /*reuse_a=*/false, /*reuse_b=*/false);
}

// Async global->LDS 16B copy (ASYNCcnt path); falls back to load+ds_store.
__device__ __forceinline__ void cp_async_b128(void* lds, const void* g) {
#if ASYNC_OK
  __builtin_amdgcn_global_load_async_to_lds_b128((v4i*)g, (v4i*)lds, 0, 0);
#else
  *(v8h*)lds = *(const v8h*)g;
#endif
}

__device__ __forceinline__ void async_fence() {
#if ASYNC_OK
#if __has_builtin(__builtin_amdgcn_s_wait_asynccnt)
  __builtin_amdgcn_s_wait_asynccnt(0);
#else
  asm volatile("s_wait_asynccnt 0x0" ::: "memory");
#endif
#endif
}

// A-fragment (16x32 f16, MxK). Lane L holds row M = L%16;
// lanes 0-15 hold K = 0..15, lanes 16-31 hold K = 16..31 (contiguous per lane).
__device__ __forceinline__ v16h load_a_frag(const _Float16* base, int ld) {
  const int lane = threadIdx.x & 31;
  const int row  = lane & 15;
  const int kq   = (lane < 16) ? 0 : 16;
  const _Float16* p = base + row * ld + kq;
  v8h lo = *(const v8h*)(p);
  v8h hi = *(const v8h*)(p + 8);
  v16h r;
#pragma unroll
  for (int i = 0; i < 8; ++i) { r[i] = lo[i]; r[i + 8] = hi[i]; }
  return r;
}

// B-fragment (32x16 f16, KxN), source row-major [n][k] (torch Linear W[n,k],
// or K rows / V^T rows). Lane L holds column N = L%16; lanes 0-15 hold
// K = 0..15, lanes 16-31 hold K = 16..31 (contiguous per lane).
__device__ __forceinline__ v16h load_b_frag(const _Float16* base, int ld) {
  const int lane = threadIdx.x & 31;
  const int n    = lane & 15;
  const int koff = (lane < 16) ? 0 : 16;
  const _Float16* p = base + n * ld + koff;
  v8h lo = *(const v8h*)(p);
  v8h hi = *(const v8h*)(p + 8);
  v16h r;
#pragma unroll
  for (int i = 0; i < 8; ++i) { r[i] = lo[i]; r[i + 8] = hi[i]; }
  return r;
}

// ---------------------------------------------------------------------------
// f32 -> f16 conversion
// ---------------------------------------------------------------------------
__global__ void f32_to_f16_kernel(const float* __restrict__ in,
                                  _Float16* __restrict__ out, size_t n) {
  size_t i = (size_t)blockIdx.x * blockDim.x + threadIdx.x;
  if (i < n) out[i] = (_Float16)in[i];
}

// ---------------------------------------------------------------------------
// GEMM: out[M,N] = A[M,K] @ W[N,K]^T + bias  (optional ReLU; f32/f16 out)
// 128 threads = 4 waves (2x2); wave tile 64x64 (16 WMMA accumulators),
// workgroup tile 128x128. Global->LDS staging via async-to-LDS copies.
// ---------------------------------------------------------------------------
__global__ __launch_bounds__(128) void gemm_f16_kernel(
    const _Float16* __restrict__ A, const _Float16* __restrict__ W,
    const float* __restrict__ bias, float* __restrict__ outF,
    _Float16* __restrict__ outH, int M, int N, int K, int relu) {
  __shared__ __align__(16) _Float16 As[128][32];
  __shared__ __align__(16) _Float16 Bs[128][32];
  const int tid  = threadIdx.x;
  const int wave = tid >> 5, lane = tid & 31;
  const int wm = wave >> 1, wn = wave & 1;
  const int m0 = blockIdx.x * 128;
  const int n0 = blockIdx.y * 128;
  const int r0 = tid >> 2;          // 0..31
  const int c0 = (tid & 3) * 8;     // 0,8,16,24
  v8f acc[4][4] = {};

  for (int k0 = 0; k0 < K; k0 += 32) {
#pragma unroll
    for (int t = 0; t < 4; ++t) {
      const int r = r0 + t * 32;
      cp_async_b128(&As[r][c0], &A[(size_t)(m0 + r) * K + k0 + c0]);
      cp_async_b128(&Bs[r][c0], &W[(size_t)(n0 + r) * K + k0 + c0]);
    }
    if (k0 + 32 < K) {  // far prefetch of the next K-slab into GL2
      __builtin_prefetch(&A[(size_t)(m0 + r0) * K + k0 + 32 + c0], 0, 1);
      __builtin_prefetch(&W[(size_t)(n0 + r0) * K + k0 + 32 + c0], 0, 1);
    }
    async_fence();
    __syncthreads();

    v16h af[4], bf[4];
#pragma unroll
    for (int i = 0; i < 4; ++i) af[i] = load_a_frag(&As[wm * 64 + i * 16][0], 32);
#pragma unroll
    for (int j = 0; j < 4; ++j) bf[j] = load_b_frag(&Bs[wn * 64 + j * 16][0], 32);
#pragma unroll
    for (int i = 0; i < 4; ++i)
#pragma unroll
      for (int j = 0; j < 4; ++j)
        acc[i][j] = wmma_f16(af[i], bf[j], acc[i][j]);
    __syncthreads();
  }

#pragma unroll
  for (int mi = 0; mi < 4; ++mi) {
#pragma unroll
    for (int ni = 0; ni < 4; ++ni) {
      const int mbase = m0 + wm * 64 + mi * 16 + ((lane < 16) ? 0 : 8);
      const int col   = n0 + wn * 64 + ni * 16 + (lane & 15);
      const float bv  = bias ? bias[col] : 0.0f;
#pragma unroll
      for (int r = 0; r < 8; ++r) {
        float v = acc[mi][ni][r] + bv;
        if (relu) v = fmaxf(v, 0.0f);
        size_t idx = (size_t)(mbase + r) * N + col;
        if (outF) outF[idx] = v;
        if (outH) outH[idx] = (_Float16)v;
      }
    }
  }
}

// ---------------------------------------------------------------------------
// V transpose: VT[b][h][d][s] = V[b*SEQ+s][h*64+d]
// ---------------------------------------------------------------------------
__global__ void transpose_v_kernel(const _Float16* __restrict__ V,
                                   _Float16* __restrict__ VT) {
  size_t i = (size_t)blockIdx.x * blockDim.x + threadIdx.x;
  int s = (int)(i & (SEQ - 1));
  size_t j = i >> 11;
  int d = (int)(j & (HEAD_DIM - 1)); j >>= 6;
  int h = (int)(j & (N_HEAD - 1));
  int b = (int)(j >> 4);
  VT[i] = V[((size_t)(b * SEQ + s)) * D_MODEL + h * HEAD_DIM + d];
}

// ---------------------------------------------------------------------------
// Flash attention: grid = B*H*(SEQ/64), block = 128 (4 waves, 16 q-rows each)
// ---------------------------------------------------------------------------
__global__ __launch_bounds__(128) void flash_attn_kernel(
    const _Float16* __restrict__ Q, const _Float16* __restrict__ Km,
    const _Float16* __restrict__ VT, _Float16* __restrict__ O) {
  __shared__ __align__(16) _Float16 Ps[4][16][32];  // per-wave P staging
  const int tid = threadIdx.x, wave = tid >> 5, lane = tid & 31;
  const int bid = blockIdx.x;
  const int qt = bid & 31;
  const int h  = (bid >> 5) & 15;
  const int b  = bid >> 9;
  const int q0 = qt * 64 + wave * 16;
  const size_t rowbase = (size_t)b * SEQ;
  const _Float16* Qbase = Q + (rowbase + q0) * D_MODEL + h * HEAD_DIM;
  const _Float16* Kbase = Km + rowbase * D_MODEL + h * HEAD_DIM;
  const _Float16* Vbase = VT + ((size_t)(b * N_HEAD + h)) * HEAD_DIM * SEQ;

  v16h qa0 = load_a_frag(Qbase, D_MODEL);       // d = 0..31
  v16h qa1 = load_a_frag(Qbase + 32, D_MODEL);  // d = 32..63
  float mrow[8], lrow[8];
#pragma unroll
  for (int r = 0; r < 8; ++r) { mrow[r] = -3.0e38f; lrow[r] = 0.0f; }
  v8f o0 = {}, o1 = {}, o2 = {}, o3 = {};

  for (int kb = 0; kb < SEQ; kb += 32) {
    // scores for 32 keys: two 16x16 tiles, K-dim = head_dim split 32+32
    v8f s0 = {}, s1 = {};
    s0 = wmma_f16(qa0, load_b_frag(Kbase + (size_t)kb * D_MODEL, D_MODEL), s0);
    s0 = wmma_f16(qa1, load_b_frag(Kbase + (size_t)kb * D_MODEL + 32, D_MODEL), s0);
    s1 = wmma_f16(qa0, load_b_frag(Kbase + (size_t)(kb + 16) * D_MODEL, D_MODEL), s1);
    s1 = wmma_f16(qa1, load_b_frag(Kbase + (size_t)(kb + 16) * D_MODEL + 32, D_MODEL), s1);

    float sc[8];
#pragma unroll
    for (int r = 0; r < 8; ++r) {
      float a = s0[r] * 0.125f;   // 1/sqrt(64)
      float e = s1[r] * 0.125f;
      float t = fmaxf(a, e);      // row-wise max across the 16-lane half
      t = fmaxf(t, __shfl_xor(t, 1, 32));
      t = fmaxf(t, __shfl_xor(t, 2, 32));
      t = fmaxf(t, __shfl_xor(t, 4, 32));
      t = fmaxf(t, __shfl_xor(t, 8, 32));
      float mn = fmaxf(mrow[r], t);
      sc[r] = __expf(mrow[r] - mn);
      mrow[r] = mn;
      float p0 = __expf(a - mn);
      float p1 = __expf(e - mn);
      s0[r] = p0; s1[r] = p1;
      float ps = p0 + p1;
      ps += __shfl_xor(ps, 1, 32);
      ps += __shfl_xor(ps, 2, 32);
      ps += __shfl_xor(ps, 4, 32);
      ps += __shfl_xor(ps, 8, 32);
      lrow[r] = lrow[r] * sc[r] + ps;
    }
#pragma unroll
    for (int r = 0; r < 8; ++r) {
      o0[r] *= sc[r]; o1[r] *= sc[r]; o2[r] *= sc[r]; o3[r] *= sc[r];
    }

    // C-layout -> A-layout re-shuffle of P via per-wave LDS region
    __syncthreads();
    {
      const int ro = (lane < 16) ? 0 : 8;
      const int cl = lane & 15;
#pragma unroll
      for (int r = 0; r < 8; ++r) {
        Ps[wave][r + ro][cl]      = (_Float16)s0[r];
        Ps[wave][r + ro][cl + 16] = (_Float16)s1[r];
      }
    }
    __syncthreads();
    v16h pa = load_a_frag(&Ps[wave][0][0], 32);
    o0 = wmma_f16(pa, load_b_frag(Vbase + (size_t)0  * SEQ + kb, SEQ), o0);
    o1 = wmma_f16(pa, load_b_frag(Vbase + (size_t)16 * SEQ + kb, SEQ), o1);
    o2 = wmma_f16(pa, load_b_frag(Vbase + (size_t)32 * SEQ + kb, SEQ), o2);
    o3 = wmma_f16(pa, load_b_frag(Vbase + (size_t)48 * SEQ + kb, SEQ), o3);
  }

  {
    const int ro = (lane < 16) ? 0 : 8;
    const int cl = lane & 15;
#pragma unroll
    for (int r = 0; r < 8; ++r) {
      float inv = 1.0f / lrow[r];
      size_t cbase = (rowbase + q0 + r + ro) * D_MODEL + h * HEAD_DIM + cl;
      O[cbase +  0] = (_Float16)(o0[r] * inv);
      O[cbase + 16] = (_Float16)(o1[r] * inv);
      O[cbase + 32] = (_Float16)(o2[r] * inv);
      O[cbase + 48] = (_Float16)(o3[r] * inv);
    }
  }
}

// ---------------------------------------------------------------------------
// out = LayerNorm(A + B) * gamma + beta  (block per row of D_MODEL)
// ---------------------------------------------------------------------------
__global__ __launch_bounds__(256) void add_ln_kernel(
    const float* __restrict__ A, const float* __restrict__ Bv,
    const float* __restrict__ gamma, const float* __restrict__ beta,
    float* __restrict__ outF, _Float16* __restrict__ outH) {
  __shared__ float red[256];
  const int row = blockIdx.x, tid = threadIdx.x;
  const size_t base = (size_t)row * D_MODEL;
  float v[4], s = 0.0f;
#pragma unroll
  for (int i = 0; i < 4; ++i) {
    int c = tid * 4 + i;
    v[i] = A[base + c] + Bv[base + c];
    s += v[i];
  }
  red[tid] = s; __syncthreads();
  for (int off = 128; off > 0; off >>= 1) {
    if (tid < off) red[tid] += red[tid + off];
    __syncthreads();
  }
  float mu = red[0] * (1.0f / D_MODEL);
  __syncthreads();
  float s2 = 0.0f;
#pragma unroll
  for (int i = 0; i < 4; ++i) { float d = v[i] - mu; v[i] = d; s2 += d * d; }
  red[tid] = s2; __syncthreads();
  for (int off = 128; off > 0; off >>= 1) {
    if (tid < off) red[tid] += red[tid + off];
    __syncthreads();
  }
  float inv = rsqrtf(red[0] * (1.0f / D_MODEL) + 1e-5f);
#pragma unroll
  for (int i = 0; i < 4; ++i) {
    int c = tid * 4 + i;
    float y = v[i] * inv * gamma[c] + beta[c];
    outF[base + c] = y;
    if (outH) outH[base + c] = (_Float16)y;
  }
}

// ---------------------------------------------------------------------------
extern "C" void kernel_launch(void* const* d_in, const int* in_sizes, int n_in,
                              void* d_out, int out_size, void* d_ws, size_t ws_size,
                              hipStream_t stream) {
  (void)in_sizes; (void)n_in; (void)out_size; (void)ws_size;
  const float* src = (const float*)d_in[0];
  const float* wq  = (const float*)d_in[1];
  const float* bq  = (const float*)d_in[2];
  const float* wk  = (const float*)d_in[3];
  const float* bk  = (const float*)d_in[4];
  const float* wv  = (const float*)d_in[5];
  const float* bv  = (const float*)d_in[6];
  const float* wo  = (const float*)d_in[7];
  const float* bo  = (const float*)d_in[8];
  const float* w1  = (const float*)d_in[9];
  const float* b1  = (const float*)d_in[10];
  const float* w2  = (const float*)d_in[11];
  const float* b2  = (const float*)d_in[12];
  const float* g1  = (const float*)d_in[13];
  const float* be1 = (const float*)d_in[14];
  const float* g2  = (const float*)d_in[15];
  const float* be2 = (const float*)d_in[16];
  float* out = (float*)d_out;

  const size_t MD = (size_t)MROWS * D_MODEL;
  const size_t DD = (size_t)D_MODEL * D_MODEL;
  const size_t FD = (size_t)D_FF * D_MODEL;
  const size_t MF = (size_t)MROWS * D_FF;

  char* ws = (char*)d_ws;
  size_t off = 0;
  auto alloc = [&](size_t bytes) -> char* {
    char* p = ws + off;
    off += (bytes + 255) & ~(size_t)255;
    return p;
  };
  _Float16* Xh    = (_Float16*)alloc(MD * 2);
  _Float16* Wqh   = (_Float16*)alloc(DD * 2);
  _Float16* Wkh   = (_Float16*)alloc(DD * 2);
  _Float16* Wvh   = (_Float16*)alloc(DD * 2);
  _Float16* Woh   = (_Float16*)alloc(DD * 2);
  _Float16* W1h   = (_Float16*)alloc(FD * 2);
  _Float16* W2h   = (_Float16*)alloc(FD * 2);
  _Float16* Qh    = (_Float16*)alloc(MD * 2);
  _Float16* Kh    = (_Float16*)alloc(MD * 2);
  _Float16* Vh    = (_Float16*)alloc(MD * 2);
  _Float16* VTh   = (_Float16*)alloc(MD * 2);
  _Float16* Oh    = (_Float16*)alloc(MD * 2);
  _Float16* x1h   = (_Float16*)alloc(MD * 2);
  _Float16* Hh    = (_Float16*)alloc(MF * 2);
  float*    attnP = (float*)alloc(MD * 4);
  float*    x1f   = (float*)alloc(MD * 4);
  float*    ff2   = (float*)alloc(MD * 4);

  auto cvt = [&](const float* in, _Float16* o, size_t n) {
    f32_to_f16_kernel<<<dim3((unsigned)((n + 255) / 256)), dim3(256), 0, stream>>>(in, o, n);
  };
  cvt(src, Xh, MD);
  cvt(wq, Wqh, DD); cvt(wk, Wkh, DD); cvt(wv, Wvh, DD); cvt(wo, Woh, DD);
  cvt(w1, W1h, FD); cvt(w2, W2h, FD);

  dim3 gblk(128);
  dim3 gD(MROWS / 128, D_MODEL / 128);
  dim3 gF(MROWS / 128, D_FF / 128);

  gemm_f16_kernel<<<gD, gblk, 0, stream>>>(Xh, Wqh, bq, nullptr, Qh, MROWS, D_MODEL, D_MODEL, 0);
  gemm_f16_kernel<<<gD, gblk, 0, stream>>>(Xh, Wkh, bk, nullptr, Kh, MROWS, D_MODEL, D_MODEL, 0);
  gemm_f16_kernel<<<gD, gblk, 0, stream>>>(Xh, Wvh, bv, nullptr, Vh, MROWS, D_MODEL, D_MODEL, 0);
  transpose_v_kernel<<<dim3((unsigned)(MD / 256)), dim3(256), 0, stream>>>(Vh, VTh);
  flash_attn_kernel<<<dim3(BATCH * N_HEAD * (SEQ / 64)), dim3(128), 0, stream>>>(Qh, Kh, VTh, Oh);
  gemm_f16_kernel<<<gD, gblk, 0, stream>>>(Oh, Woh, bo, attnP, nullptr, MROWS, D_MODEL, D_MODEL, 0);
  add_ln_kernel<<<dim3(MROWS), dim3(256), 0, stream>>>(src, attnP, g1, be1, x1f, x1h);
  gemm_f16_kernel<<<gF, gblk, 0, stream>>>(x1h, W1h, b1, nullptr, Hh, MROWS, D_FF, D_MODEL, 1);
  gemm_f16_kernel<<<gD, gblk, 0, stream>>>(Hh, W2h, b2, ff2, nullptr, MROWS, D_MODEL, D_FF, 0);
  add_ln_kernel<<<dim3(MROWS), dim3(256), 0, stream>>>(x1f, ff2, g2, be2, out, nullptr);
}
